// MultiHeadAttentionWithALiBi_13778255085639
// MI455X (gfx1250) — matrix operand order repaired and checked by reading, not compiled, not run on hardware
//
#include <hip/hip_runtime.h>

#define B_  4
#define S_  2048
#define D_  512
#define H_  8
#define HD_ 64

typedef __attribute__((ext_vector_type(16))) __bf16 v16bf;
typedef __attribute__((ext_vector_type(8)))  __bf16 v8bf;
typedef __attribute__((ext_vector_type(8)))  float  v8f;

// Build a 16-element bf16 WMMA fragment from two contiguous 8-element (16B) chunks.
__device__ __forceinline__ v16bf load_frag(const __bf16* base, int off0, int off1) {
    v8bf lo = *(const v8bf*)(base + off0);
    v8bf hi = *(const v8bf*)(base + off1);
    v16bf r;
#pragma unroll
    for (int i = 0; i < 8; ++i) { r[i] = lo[i]; r[i + 8] = hi[i]; }
    return r;
}

__device__ __forceinline__ v8f bf_wmma(v16bf a, v16bf b, v8f c) {
    return __builtin_amdgcn_wmma_f32_16x16x32_bf16(
        /*neg_a=*/false, a, /*neg_b=*/false, b,
        /*c_mod=*/(short)0, c, /*reuse_a=*/false, /*reuse_b=*/false);
}

// ---------------- fp32 -> bf16 conversion ----------------
__global__ void f2bf_kernel(const float* __restrict__ src, __bf16* __restrict__ dst, int n) {
    int i = blockIdx.x * blockDim.x + threadIdx.x;
    if (i < n) dst[i] = (__bf16)src[i];
}

// ---------------- GEMM: Y[M,N] = A[M,K] * W[N,K]^T + bias ----------------
// One wave computes one 16x16 tile. A-layout: lane m=L&15, k = 8*(L>>4) + {0..7, 16..23} per 32-step.
// B-layout: lane n=L&15, k = 8*(L>>4) + {0..7, 16..23} (same k order as A).
// MODE 0: bf16 row-major out.  MODE 1: f32 row-major out (non-temporal, final result).
// MODE 2: bf16 out scattered into V^T layout (b,h,d,s) for the P*V B-operand.
template <int MODE>
__global__ void gemm_wmma(const __bf16* __restrict__ A, const __bf16* __restrict__ W,
                          const float* __restrict__ bias,
                          float* __restrict__ Yf, __bf16* __restrict__ Yb,
                          int M, int N, int K) {
    int wave = threadIdx.x >> 5;
    int lane = threadIdx.x & 31;
    int tile = blockIdx.x * 4 + wave;
    int tn = tile % (N >> 4);
    int tm = tile / (N >> 4);
    int nl = lane & 15;
    int g  = lane >> 4;

    const __bf16* pa = A + (size_t)(tm * 16 + nl) * K;
    const __bf16* pb = W + (size_t)(tn * 16 + nl) * K;

    v8f c = {};
    for (int kk = 0; kk < K; kk += 32) {
        v16bf af = load_frag(pa, kk + g * 8,  kk + 16 + g * 8);
        v16bf bf = load_frag(pb, kk + g * 8,  kk + 16 + g * 8);
        c = bf_wmma(af, bf, c);
    }

    float bv = bias[tn * 16 + nl];
    int col = tn * 16 + nl;
#pragma unroll
    for (int r = 0; r < 8; ++r) {
        int row = tm * 16 + r + 8 * g;
        float v = c[r] + bv;
        if (MODE == 1) {
            __builtin_nontemporal_store(v, &Yf[(size_t)row * N + col]);
        } else if (MODE == 0) {
            Yb[(size_t)row * N + col] = (__bf16)v;
        } else {
            // V^T scatter: row = b*S + s, col = h*HD + d -> Vt[((b*H+h)*HD+d)*S + s]
            int b = row >> 11, s = row & (S_ - 1);
            int h = col >> 6,  d = col & (HD_ - 1);
            Yb[((size_t)(b * H_ + h) * HD_ + d) * S_ + s] = (__bf16)v;
        }
    }
}

// ---------------- Attention: scores + ALiBi + mask + softmax + P*V ----------------
// One wave per (b, h, 16-query) tile. Pass 1: online row max/sum.
// Pass 2: recompute scores (free vs HBM), stream normalized attn out (NT stores),
// transpose P through LDS, WMMA-accumulate P*V.
__global__ void attn_wmma(const __bf16* __restrict__ Qb, const __bf16* __restrict__ Kb,
                          const __bf16* __restrict__ Vt, const int* __restrict__ amask,
                          float* __restrict__ attn, __bf16* __restrict__ Hout) {
    __shared__ __bf16 p_lds[4][16 * 32];

    int wave = threadIdx.x >> 5;
    int lane = threadIdx.x & 31;
    int tile = blockIdx.x * 4 + wave;
    const int QT = S_ / 16;                 // 128 query tiles per (b,h)
    int qt = tile % QT;
    int bh = tile / QT;
    int h  = bh % H_;
    int b  = bh / H_;
    int nl = lane & 15;
    int g  = lane >> 4;

    const float scale = 0.125f;             // 1/sqrt(64)
    float slope = exp2f(-(float)(h + 1) * (8.0f / (float)H_));

    // Q A-fragments for this query tile (HD=64 -> two K=32 fragments)
    const __bf16* qbase = Qb + (size_t)(b * S_ + qt * 16 + nl) * D_ + h * HD_;
    v16bf qa0 = load_frag(qbase, g * 8,      16 + g * 8);
    v16bf qa1 = load_frag(qbase, 32 + g * 8, 48 + g * 8);

    const __bf16* khead = Kb + (size_t)(b * S_) * D_ + h * HD_;

    float rm[8], rs[8];
#pragma unroll
    for (int r = 0; r < 8; ++r) { rm[r] = -__builtin_inff(); rs[r] = 0.f; }

    // ---- pass 1: online softmax statistics ----
    for (int kt = 0; kt < QT; ++kt) {
        int key = kt * 16 + nl;
        const __bf16* kbase = khead + (size_t)key * D_;
        __builtin_prefetch(kbase + 16 * D_, 0, 1);   // next tile's K rows
        v16bf kb0 = load_frag(kbase, g * 8,      16 + g * 8);
        v16bf kb1 = load_frag(kbase, 32 + g * 8, 48 + g * 8);
        v8f c = {};
        c = bf_wmma(qa0, kb0, c);
        c = bf_wmma(qa1, kb1, c);
        int mv = amask[b * S_ + key];
#pragma unroll
        for (int r = 0; r < 8; ++r) {
            int q = qt * 16 + r + 8 * g;
            float x = c[r] * scale - slope * fabsf((float)(q - key));
            if (mv == 0) x = -__builtin_inff();
            float nm = fmaxf(rm[r], x);
            rs[r] = rs[r] * __expf(rm[r] - nm) + __expf(x - nm);
            rm[r] = nm;
        }
    }
    // merge across the 16 lanes that share each row (xor<16 stays in-half on wave32)
#pragma unroll
    for (int r = 0; r < 8; ++r) {
#pragma unroll
        for (int off = 1; off < 16; off <<= 1) {
            float om = __shfl_xor(rm[r], off, 32);
            float os = __shfl_xor(rs[r], off, 32);
            float nm = fmaxf(rm[r], om);
            rs[r] = rs[r] * __expf(rm[r] - nm) + os * __expf(om - nm);
            rm[r] = nm;
        }
    }
    float inv[8];
#pragma unroll
    for (int r = 0; r < 8; ++r) inv[r] = 1.0f / rs[r];

    v8f oacc[4];
#pragma unroll
    for (int nt = 0; nt < 4; ++nt) oacc[nt] = (v8f){};

    // ---- pass 2: recompute scores, emit attn (NT), accumulate P*V ----
    float* arow = attn + (size_t)(b * H_ + h) * S_ * S_;
    for (int kt = 0; kt < QT; ++kt) {
        int key = kt * 16 + nl;
        const __bf16* kbase = khead + (size_t)key * D_;
        v16bf kb0 = load_frag(kbase, g * 8,      16 + g * 8);
        v16bf kb1 = load_frag(kbase, 32 + g * 8, 48 + g * 8);
        v8f c = {};
        c = bf_wmma(qa0, kb0, c);
        c = bf_wmma(qa1, kb1, c);
        int mv = amask[b * S_ + key];
#pragma unroll
        for (int r = 0; r < 8; ++r) {
            int q = qt * 16 + r + 8 * g;
            float x = c[r] * scale - slope * fabsf((float)(q - key));
            float p = (mv == 0) ? 0.f : __expf(x - rm[r]) * inv[r];
            // 537MB streaming output, never re-read: bypass L2 retention (TH=NT)
            __builtin_nontemporal_store(p, &arow[(size_t)q * S_ + key]);
            p_lds[wave][(r + 8 * g) * 32 + (kt & 1) * 16 + nl] = (__bf16)p;
        }
        if (kt & 1) {
            __syncthreads();
            // transpose P tile (16x32) out of LDS into A-fragment layout
            const __bf16* pl = &p_lds[wave][nl * 32];
            v16bf pa = load_frag(pl, g * 8, 16 + g * 8);
            int k0 = (kt - 1) * 16;
#pragma unroll
            for (int nt = 0; nt < 4; ++nt) {
                const __bf16* vb =
                    Vt + ((size_t)(b * H_ + h) * HD_ + nt * 16 + nl) * S_ + k0;
                v16bf bf = load_frag(vb, g * 8, 16 + g * 8);
                oacc[nt] = bf_wmma(pa, bf, oacc[nt]);
            }
            __syncthreads();
        }
    }

    // store per-head output (bf16) for the final O-projection GEMM
#pragma unroll
    for (int nt = 0; nt < 4; ++nt) {
#pragma unroll
        for (int r = 0; r < 8; ++r) {
            int q = qt * 16 + r + 8 * g;
            Hout[(size_t)(b * S_ + q) * D_ + h * HD_ + nt * 16 + nl] = (__bf16)oacc[nt][r];
        }
    }
}

extern "C" void kernel_launch(void* const* d_in, const int* in_sizes, int n_in,
                              void* d_out, int out_size, void* d_ws, size_t ws_size,
                              hipStream_t stream) {
    const float* x  = (const float*)d_in[0];
    const int*   am = (const int*)  d_in[1];
    const float* Wq = (const float*)d_in[2];
    const float* bq = (const float*)d_in[3];
    const float* Wk = (const float*)d_in[4];
    const float* bk = (const float*)d_in[5];
    const float* Wv = (const float*)d_in[6];
    const float* bv = (const float*)d_in[7];
    const float* Wo = (const float*)d_in[8];
    const float* bo = (const float*)d_in[9];

    float* out  = (float*)d_out;
    float* attn = out + (size_t)B_ * S_ * D_;   // outputs concatenated: (out, attn)

    const size_t nX = (size_t)B_ * S_ * D_;     // 4,194,304 elements
    const size_t nW = (size_t)D_ * D_;          // 262,144 elements

    char* w = (char*)d_ws;
    __bf16* xb  = (__bf16*)w; w += nX * 2;
    __bf16* Wqb = (__bf16*)w; w += nW * 2;
    __bf16* Wkb = (__bf16*)w; w += nW * 2;
    __bf16* Wvb = (__bf16*)w; w += nW * 2;
    __bf16* Wob = (__bf16*)w; w += nW * 2;
    __bf16* Qb  = (__bf16*)w; w += nX * 2;
    __bf16* Kb  = (__bf16*)w; w += nX * 2;
    __bf16* Vt  = (__bf16*)w; w += nX * 2;
    __bf16* Ho  = (__bf16*)w; w += nX * 2;

    // fp32 -> bf16
    f2bf_kernel<<<(nX + 255) / 256, 256, 0, stream>>>(x,  xb,  (int)nX);
    f2bf_kernel<<<(nW + 255) / 256, 256, 0, stream>>>(Wq, Wqb, (int)nW);
    f2bf_kernel<<<(nW + 255) / 256, 256, 0, stream>>>(Wk, Wkb, (int)nW);
    f2bf_kernel<<<(nW + 255) / 256, 256, 0, stream>>>(Wv, Wvb, (int)nW);
    f2bf_kernel<<<(nW + 255) / 256, 256, 0, stream>>>(Wo, Wob, (int)nW);

    const int M = B_ * S_, N = D_, K = D_;
    const int tiles = (M / 16) * (N / 16);      // 16384 -> 4096 blocks of 4 waves
    gemm_wmma<0><<<tiles / 4, 128, 0, stream>>>(xb, Wqb, bq, nullptr, Qb, M, N, K);
    gemm_wmma<0><<<tiles / 4, 128, 0, stream>>>(xb, Wkb, bk, nullptr, Kb, M, N, K);
    gemm_wmma<2><<<tiles / 4, 128, 0, stream>>>(xb, Wvb, bv, nullptr, Vt, M, N, K);  // fused V^T

    const int atiles = B_ * H_ * (S_ / 16);     // 4096 -> 1024 blocks of 4 waves
    attn_wmma<<<atiles / 4, 128, 0, stream>>>(Qb, Kb, Vt, am, attn, Ho);

    gemm_wmma<1><<<tiles / 4, 128, 0, stream>>>(Ho, Wob, bo, out, nullptr, M, N, K);
}
